// HomoMSELoss_63728724738202
// MI455X (gfx1250) — matrix operand order repaired and checked by reading, not compiled, not run on hardware
//
#include <hip/hip_runtime.h>
#include <hip/hip_bf16.h>

#define BB 32
#define HH 480
#define WW 640
#define NN (HH * WW)
#define TOPK 1024
#define CAP 16384
#define NMS_TH 0.3f
#define TILES_X (WW / 16)           // 40
#define TILES_Y (HH / 16)           // 30
#define TILES_PER_IMG (TILES_X * TILES_Y)  // 1200, divisible by 8

typedef __attribute__((ext_vector_type(16))) _Float16 v16h;
typedef __attribute__((ext_vector_type(8)))  float    v8f;

// 1-D normalized gaussian factor (k=5, sigma=2); outer(g,g)/sum == g1 (x) g1.
// Branchless select chain -> v_cmp/v_cndmask with inline f16 constants.
__device__ __forceinline__ _Float16 gaussh(int d) {
  _Float16 r = (_Float16)0.0f;
  r = (d == 0 || d == 4) ? (_Float16)0.15246914f : r;
  r = (d == 1 || d == 3) ? (_Float16)0.22184130f : r;
  r = (d == 2)           ? (_Float16)0.25137913f : r;
  return r;
}

// ---------------------------------------------------------------------------
// Kernel 1: homography warp of score2 + raw visibility mask (bilinear, zero pad)
// ---------------------------------------------------------------------------
__global__ void k_warp(const float* __restrict__ s2, const float* __restrict__ homo,
                       float* __restrict__ w2, float* __restrict__ mraw) {
  int b = blockIdx.y;
  int i = blockIdx.x * blockDim.x + threadIdx.x;
  if (i >= NN) return;
  const float* h = homo + b * 9;
  float h00 = h[0], h01 = h[1], h02 = h[2];
  float h10 = h[3], h11 = h[4], h12 = h[5];
  float h20 = h[6], h21 = h[7], h22 = h[8];
  int y = i / WW, x = i - y * WW;
  float fx = (float)x, fy = (float)y;
  float hz = h20 * fx + h21 * fy + h22 + 1e-8f;
  float px = (h00 * fx + h01 * fy + h02) / hz;
  float py = (h10 * fx + h11 * fy + h12) / hz;
  float x0f = floorf(px), y0f = floorf(py);
  float ax = px - x0f, ay = py - y0f;
  int ix = (int)x0f, iy = (int)y0f;
  const float* img = s2 + (size_t)b * NN;
  float v[4], o[4];
  const int dxs[4] = {0, 1, 0, 1};
  const int dys[4] = {0, 0, 1, 1};
#pragma unroll
  for (int c = 0; c < 4; ++c) {
    int cx = ix + dxs[c], cy = iy + dys[c];
    bool ok = (cx >= 0) && (cx < WW) && (cy >= 0) && (cy < HH);
    v[c] = ok ? img[cy * WW + cx] : 0.0f;
    o[c] = ok ? 1.0f : 0.0f;
  }
  float w00 = (1.0f - ax) * (1.0f - ay), w01 = ax * (1.0f - ay);
  float w10 = (1.0f - ax) * ay,          w11 = ax * ay;
  float wv  = v[0] * w00 + v[1] * w01 + v[2] * w10 + v[3] * w11;
  float vis = o[0] * w00 + o[1] * w01 + o[2] * w10 + o[3] * w11;
  w2[(size_t)b * NN + i]   = wv;
  mraw[(size_t)b * NN + i] = (vis > 0.0f) ? 1.0f : 0.0f;
}

// ---------------------------------------------------------------------------
// Kernel 2: 3x3 erosion (zero border) + norm = sum(vis)
// ---------------------------------------------------------------------------
__global__ void k_erode(const float* __restrict__ mraw, float* __restrict__ vis,
                        float* __restrict__ norm_acc) {
  __shared__ float red[256];
  int b = blockIdx.y;
  int i = blockIdx.x * blockDim.x + threadIdx.x;
  float mn = 0.0f;
  if (i < NN) {
    int y = i / WW, x = i - y * WW;
    mn = 1e30f;
    for (int dy = -1; dy <= 1; ++dy)
      for (int dx = -1; dx <= 1; ++dx) {
        int yy = y + dy, xx = x + dx;
        float val = (yy >= 0 && yy < HH && xx >= 0 && xx < WW)
                        ? mraw[(size_t)b * NN + yy * WW + xx] : 0.0f;
        mn = fminf(mn, val);
      }
    vis[(size_t)b * NN + i] = mn;
  }
  red[threadIdx.x] = mn;
  __syncthreads();
  for (int s = 128; s > 0; s >>= 1) {
    if ((int)threadIdx.x < s) red[threadIdx.x] += red[threadIdx.x + s];
    __syncthreads();
  }
  if (threadIdx.x == 0) atomicAdd(norm_acc, red[0]);
}

// ---------------------------------------------------------------------------
// Kernel 3: 5x5 NMS + threshold -> candidate list (value, flat idx) per image
// ---------------------------------------------------------------------------
__global__ void k_nms(const float* __restrict__ map, const float* __restrict__ vis,
                      float* __restrict__ cvals, int* __restrict__ cidx,
                      int* __restrict__ ccnt) {
  int b = blockIdx.y;
  int i = blockIdx.x * blockDim.x + threadIdx.x;
  if (i >= NN) return;
  const float* m  = map + (size_t)b * NN;
  const float* vv = vis + (size_t)b * NN;
  float s = m[i] * vv[i];
  if (s <= NMS_TH) return;
  int y = i / WW, x = i - y * WW;
  bool ismax = true;
  for (int dy = -2; dy <= 2 && ismax; ++dy) {
    int yy = y + dy;
    if (yy < 0 || yy >= HH) continue;
    for (int dx = -2; dx <= 2; ++dx) {
      int xx = x + dx;
      if (xx < 0 || xx >= WW) continue;
      if (m[yy * WW + xx] * vv[yy * WW + xx] > s) { ismax = false; break; }
    }
  }
  if (!ismax) return;
  int pos = atomicAdd(&ccnt[b], 1);
  if (pos < CAP) {
    cvals[(size_t)b * CAP + pos] = s;
    cidx[(size_t)b * CAP + pos]  = i;
  }
}

// ---------------------------------------------------------------------------
// Kernel 4: per-image top-1024 (radix-select on float bits + bitonic sort)
// map 0 -> keypoint coords out, map 1 -> scatter into dense gt buffer
// ---------------------------------------------------------------------------
__global__ void __launch_bounds__(1024)
k_topk(const float* __restrict__ cv1, const int* __restrict__ ci1,
       const int* __restrict__ cn1,
       const float* __restrict__ cv2, const int* __restrict__ ci2,
       const int* __restrict__ cn2,
       float* __restrict__ kp_out, float* __restrict__ gt_sparse) {
  int b = blockIdx.x;
  int mapi = blockIdx.y;
  const float* vals = (mapi == 0 ? cv1 : cv2) + (size_t)b * CAP;
  const int*   idxs = (mapi == 0 ? ci1 : ci2) + (size_t)b * CAP;
  int n = (mapi == 0 ? cn1 : cn2)[b];
  n = n < CAP ? n : CAP;
  int tid = threadIdx.x;

  __shared__ int   hist[256];
  __shared__ float sv[TOPK];
  __shared__ int   si[TOPK];
  __shared__ int   scnt;
  __shared__ unsigned s_prefix;
  __shared__ int   s_remaining;

  unsigned Tbits = 0;
  if (n > TOPK) {
    if (tid == 0) { s_prefix = 0u; s_remaining = TOPK; }
    __syncthreads();
    for (int shift = 24; shift >= 0; shift -= 8) {
      for (int hb = tid; hb < 256; hb += blockDim.x) hist[hb] = 0;
      __syncthreads();
      unsigned prefix = s_prefix;
      for (int j = tid; j < n; j += blockDim.x) {
        unsigned bits = __float_as_uint(vals[j]);  // all > 0.3 -> positive, monotonic
        bool match = (shift == 24) ||
                     ((bits >> (shift + 8)) == (prefix >> (shift + 8)));
        if (match) atomicAdd(&hist[(bits >> shift) & 255], 1);
      }
      __syncthreads();
      if (tid == 0) {
        int rem = s_remaining, above = 0, v = 255;
        for (; v >= 0; --v) {
          if (above + hist[v] >= rem) break;
          above += hist[v];
        }
        if (v < 0) v = 0;
        s_prefix = prefix | ((unsigned)v << shift);
        s_remaining = rem - above;
      }
      __syncthreads();
    }
    Tbits = s_prefix;
  }
  if (tid == 0) scnt = 0;
  __syncthreads();
  if (n <= TOPK) {
    for (int j = tid; j < n; j += blockDim.x) {
      int p = atomicAdd(&scnt, 1);
      sv[p] = vals[j]; si[p] = idxs[j];
    }
  } else {
    for (int j = tid; j < n; j += blockDim.x) {       // strictly greater first
      unsigned bits = __float_as_uint(vals[j]);
      if (bits > Tbits) {
        int p = atomicAdd(&scnt, 1);
        if (p < TOPK) { sv[p] = vals[j]; si[p] = idxs[j]; }
      }
    }
    __syncthreads();
    for (int j = tid; j < n; j += blockDim.x) {       // then ties at threshold
      unsigned bits = __float_as_uint(vals[j]);
      if (bits == Tbits) {
        int p = atomicAdd(&scnt, 1);
        if (p < TOPK) { sv[p] = vals[j]; si[p] = idxs[j]; }
      }
    }
  }
  __syncthreads();
  int filled = scnt < TOPK ? scnt : TOPK;
  for (int p = filled + tid; p < TOPK; p += blockDim.x) { sv[p] = 0.0f; si[p] = p; }
  __syncthreads();
  // bitonic sort, descending by value (ties: ascending index) — matches top_k order
  for (int k = 2; k <= TOPK; k <<= 1) {
    for (int j = k >> 1; j > 0; j >>= 1) {
      int ixj = tid ^ j;
      if (ixj > tid) {
        float va = sv[tid], vb = sv[ixj];
        int ia = si[tid], ib = si[ixj];
        bool before = (va > vb) || (va == vb && ia < ib);
        bool descSeg = ((tid & k) == 0);
        bool sw = descSeg ? (!before) : before;
        if (sw) { sv[tid] = vb; sv[ixj] = va; si[tid] = ib; si[ixj] = ia; }
      }
      __syncthreads();
    }
  }
  if (mapi == 0) {
    int id = si[tid];
    int yy = id / WW;
    kp_out[((size_t)b * TOPK + tid) * 2 + 0] = (float)yy;
    kp_out[((size_t)b * TOPK + tid) * 2 + 1] = (float)(id - yy * WW);
  } else {
    if (sv[tid] > 0.0f) gt_sparse[(size_t)b * NN + si[tid]] = sv[tid];
  }
}

// ---------------------------------------------------------------------------
// Kernel 5: fused 5x5 gaussian blur (separable, 3 WMMAs) + squared-error loss.
// One wave per 16x16 tile, 8 waves (8 tiles) per 256-thread block.
//   WMMA#1: h-blur rows y0-2..y0+13   O1 = I(16x32, cols x0-4..x0+27) * Gh
//   WMMA#2: h-blur rows y0+14..y0+17  (only M<4 used)
//   stage 20x16 h-blurred tile in LDS (stride 17 -> conflict-free)
//   WMMA#3: v-blur   O = Gv(16x32 band) * T(32x16 from LDS)
//   loss partial = sum((score1*vis - O)^2) -> wave shfl reduce -> atomicAdd
// Gh[k][n] = g1[k-n-2] (K window shifted by -4 so 8-wide K groups are 16B
// aligned: interior tiles take uniform b128 fast path). Gv[m][k] = g1[k-m].
// Intra-wave LDS handoff needs only the CDNA5 split DS counter, not a
// workgroup barrier: s_wait_dscnt 0.
// ---------------------------------------------------------------------------
__global__ void __launch_bounds__(256)
k_blur_loss(const float* __restrict__ in, const float* __restrict__ score1,
            const float* __restrict__ vis, float* __restrict__ sumsq) {
  __shared__ float Tsm[8][20][17];
  int lane = threadIdx.x;            // 0..31 (wave32)
  int wid  = threadIdx.y;            // 0..7
  int b = blockIdx.y;
  int tile = blockIdx.x * 8 + wid;
  int ty = tile / TILES_X, tx = tile - ty * TILES_X;
  int x0 = tx * 16, y0 = ty * 16;
  int hf = lane >> 4, m = lane & 15;
  const float* src = in + (size_t)b * NN;
  const float* s1  = score1 + (size_t)b * NN;
  const float* vv  = vis + (size_t)b * NN;

  // warm the loss operands while the matrix pipe works (global_prefetch_b8)
  __builtin_prefetch(s1 + (y0 + m) * WW + x0, 0, 1);
  __builtin_prefetch(vv + (y0 + m) * WW + x0, 0, 1);

  // H-pass coefficient fragment: B[k][n] = g1[k-n-2], lane holds N=m, K halves {0-7,16-23} / {8-15,24-31}
  v16h bh;
#pragma unroll
  for (int t = 0; t < 16; ++t) bh[t] = gaussh(((t < 8) ? hf * 8 + t : 8 + hf * 8 + t) - m - 2);

  // A fragments: lane holds M=m, K halves {0-7,16-23} / {8-15,24-31}
  int start = x0 - 4 + hf * 8;       // col of K group base; 16B aligned
  int r1 = y0 - 2 + m;
  int r2 = y0 + 14 + m;
  bool v1 = (r1 >= 0) && (r1 < HH);
  bool v2 = (m < 4) && (r2 < HH);
  bool fastx = (x0 >= 16) && (x0 <= WW - 32);   // wave-uniform
  v16h a1 = {}, a2 = {};
  if (fastx) {
    if (v1) {
      const float4* p = (const float4*)(src + r1 * WW + start);
      float4 q0 = p[0], q1 = p[1], q2 = p[4], q3 = p[5];
      a1[0] = (_Float16)q0.x; a1[1] = (_Float16)q0.y; a1[2] = (_Float16)q0.z; a1[3] = (_Float16)q0.w;
      a1[4] = (_Float16)q1.x; a1[5] = (_Float16)q1.y; a1[6] = (_Float16)q1.z; a1[7] = (_Float16)q1.w;
      a1[8]  = (_Float16)q2.x; a1[9]  = (_Float16)q2.y; a1[10] = (_Float16)q2.z; a1[11] = (_Float16)q2.w;
      a1[12] = (_Float16)q3.x; a1[13] = (_Float16)q3.y; a1[14] = (_Float16)q3.z; a1[15] = (_Float16)q3.w;
    }
    if (v2) {
      const float4* p = (const float4*)(src + r2 * WW + start);
      float4 q0 = p[0], q1 = p[1], q2 = p[4], q3 = p[5];
      a2[0] = (_Float16)q0.x; a2[1] = (_Float16)q0.y; a2[2] = (_Float16)q0.z; a2[3] = (_Float16)q0.w;
      a2[4] = (_Float16)q1.x; a2[5] = (_Float16)q1.y; a2[6] = (_Float16)q1.z; a2[7] = (_Float16)q1.w;
      a2[8]  = (_Float16)q2.x; a2[9]  = (_Float16)q2.y; a2[10] = (_Float16)q2.z; a2[11] = (_Float16)q2.w;
      a2[12] = (_Float16)q3.x; a2[13] = (_Float16)q3.y; a2[14] = (_Float16)q3.z; a2[15] = (_Float16)q3.w;
    }
  } else {
#pragma unroll
    for (int t = 0; t < 8; ++t) {
      int c0 = start + t, c1 = start + 16 + t;
      bool k0ok = (c0 >= 0) && (c0 < WW);
      bool k1ok = (c1 >= 0) && (c1 < WW);
      if (v1) {
        a1[t]     = (_Float16)(k0ok ? src[r1 * WW + c0] : 0.0f);
        a1[t + 8] = (_Float16)(k1ok ? src[r1 * WW + c1] : 0.0f);
      }
      if (v2) {
        a2[t]     = (_Float16)(k0ok ? src[r2 * WW + c0] : 0.0f);
        a2[t + 8] = (_Float16)(k1ok ? src[r2 * WW + c1] : 0.0f);
      }
    }
  }
  v8f z = {};
  v8f d1 = __builtin_amdgcn_wmma_f32_16x16x32_f16(false, a1, false, bh,
                                                  (short)0, z, false, false);
  v8f d2 = __builtin_amdgcn_wmma_f32_16x16x32_f16(false, a2, false, bh,
                                                  (short)0, z, false, false);
  // stage 20x16 h-blurred tile: d1 -> rows 0..15, d2 -> rows 16..19
#pragma unroll
  for (int r = 0; r < 8; ++r) Tsm[wid][r + 8 * hf][m] = d1[r];
  if (hf == 0) {
#pragma unroll
    for (int r = 0; r < 4; ++r) Tsm[wid][16 + r][m] = d2[r];
  }
  // producer and consumer are the same wave: split DS counter wait suffices
  asm volatile("s_wait_dscnt 0x0" ::: "memory");

  // V-pass: A = band matrix (M=m), B = staged tile columns (N=m).
  // For k >= 20 the A coefficient g1[k-m] is identically zero (k-m >= 5), so
  // those B slots may hold any finite value: wrap the row address instead of
  // guarding the load -> 16 unconditional, clause-able ds_load_b32.
  v16h av, bv;
#pragma unroll
  for (int t = 0; t < 8; ++t) {
    av[t]     = gaussh(hf * 8 + t - m);
    av[t + 8] = gaussh(16 + hf * 8 + t - m);
  }
#pragma unroll
  for (int t = 0; t < 16; ++t) {
    int k = (t < 8) ? hf * 8 + t : 8 + hf * 8 + t;
    int row = (k < 20) ? k : (k - 20);   // address select, not a guarded load
    bv[t] = (_Float16)Tsm[wid][row][m];
  }
  v8f d = __builtin_amdgcn_wmma_f32_16x16x32_f16(false, av, false, bv,
                                                 (short)0, z, false, false);
  float acc = 0.0f;
#pragma unroll
  for (int r = 0; r < 8; ++r) {
    int M = r + 8 * hf;
    int p = (y0 + M) * WW + (x0 + m);
    float diff = s1[p] * vv[p] - d[r];
    acc += diff * diff;
  }
#pragma unroll
  for (int o = 16; o > 0; o >>= 1) acc += __shfl_xor(acc, o);
  if (lane == 0) atomicAdd(sumsq, acc);
}

__global__ void k_finalize(const float* __restrict__ sumsq,
                           const float* __restrict__ norm,
                           float* __restrict__ out) {
  out[0] = sumsq[0] / norm[0];
}

// ---------------------------------------------------------------------------
extern "C" void kernel_launch(void* const* d_in, const int* in_sizes, int n_in,
                              void* d_out, int out_size, void* d_ws, size_t ws_size,
                              hipStream_t stream) {
  (void)in_sizes; (void)n_in; (void)out_size; (void)ws_size;
  const float* score1 = (const float*)d_in[0];
  const float* score2 = (const float*)d_in[1];
  const float* homo   = (const float*)d_in[2];

  float* out    = (float*)d_out;
  float* kp_out = out + 1;                               // [B,1024,2]
  float* vis    = out + 1 + (size_t)BB * TOPK * 2;       // [B,1,H,W]

  char* ws = (char*)d_ws;
  size_t off = 0;
  float* bufA = (float*)(ws + off); off += (size_t)BB * NN * sizeof(float); // w2 -> gt_sparse
  float* bufB = (float*)(ws + off); off += (size_t)BB * NN * sizeof(float); // mraw
  float* cv1  = (float*)(ws + off); off += (size_t)BB * CAP * sizeof(float);
  float* cv2  = (float*)(ws + off); off += (size_t)BB * CAP * sizeof(float);
  int*   ci1  = (int*)(ws + off);   off += (size_t)BB * CAP * sizeof(int);
  int*   ci2  = (int*)(ws + off);   off += (size_t)BB * CAP * sizeof(int);
  int*   cnts = (int*)(ws + off);   off += 2 * BB * sizeof(int); // [cnt1|cnt2]
  float* scal = (float*)(ws + off); off += 2 * sizeof(float);    // [norm|sumsq]
  int*   cnt1 = cnts;
  int*   cnt2 = cnts + BB;
  float* normp = scal;
  float* ssq   = scal + 1;

  hipMemsetAsync(cnts, 0, 2 * BB * sizeof(int), stream);
  hipMemsetAsync(scal, 0, 2 * sizeof(float), stream);

  dim3 blk(256), grid((NN + 255) / 256, BB);
  k_warp<<<grid, blk, 0, stream>>>(score2, homo, bufA, bufB);
  k_erode<<<grid, blk, 0, stream>>>(bufB, vis, normp);
  k_nms<<<grid, blk, 0, stream>>>(score1, vis, cv1, ci1, cnt1);
  k_nms<<<grid, blk, 0, stream>>>(bufA, vis, cv2, ci2, cnt2);

  // bufA (w2) fully consumed by NMS: reuse as dense gt_sparse
  hipMemsetAsync(bufA, 0, (size_t)BB * NN * sizeof(float), stream);
  k_topk<<<dim3(BB, 2), 1024, 0, stream>>>(cv1, ci1, cnt1, cv2, ci2, cnt2,
                                           kp_out, bufA);

  // fused h-blur + v-blur + loss: 8 tiles (8 waves) per block
  dim3 tgrid(TILES_PER_IMG / 8, BB);
  k_blur_loss<<<tgrid, dim3(32, 8), 0, stream>>>(bufA, score1, vis, ssq);
  k_finalize<<<1, 1, 0, stream>>>(ssq, normp, out);
}
